// GIN_Model_79680233276319
// MI455X (gfx1250) — matrix operand order repaired and checked by reading, not compiled, not run on hardware
//
#include <hip/hip_runtime.h>
#include <hip/hip_bf16.h>

constexpr int kB = 64;    // batch
constexpr int kN = 512;   // nodes
constexpr int kD = 64;    // feature dim
constexpr int kH = 64;    // hidden dim
constexpr int kL = 4;     // GIN layers
constexpr int kC = 10;    // classes
constexpr int kGD = kD * (kL + 1);  // 320

typedef __attribute__((ext_vector_type(16))) __bf16 v16bf;
typedef __attribute__((ext_vector_type(8)))  float  v8f;

union Frag {
  v16bf bf;
  unsigned int u32[8];
  unsigned short us[16];
};

// fp32 -> bf16 with round-to-nearest-even
__device__ __forceinline__ unsigned short f2bf(float f) {
  unsigned int u = __float_as_uint(f);
  u += 0x7FFFu + ((u >> 16) & 1u);
  return (unsigned short)(u >> 16);
}
__device__ __forceinline__ unsigned int pack2(float x, float y) {
  return (unsigned int)f2bf(x) | ((unsigned int)f2bf(y) << 16);
}

__device__ __forceinline__ v8f wmma_bf16(v16bf a, v16bf b, v8f c) {
  // (neg_a, A, neg_b, B, c_mod, C, reuse_a, reuse_b)
  return __builtin_amdgcn_wmma_f32_16x16x32_bf16(false, a, false, b, (short)0, c,
                                                 false, false);
}

// ---- CDNA5 async global->LDS copy (ASYNCcnt-tracked), 16 bytes ------------
// Builtin signature (from hipcc diagnostic): param0 = int4 (vector_size 16)
// pointer in global AS(1), param1 = int4 pointer in LDS AS(3), then two imms.
typedef int i128 __attribute__((vector_size(16)));
typedef __attribute__((address_space(1))) i128* gp_i128;
typedef __attribute__((address_space(3))) i128* lp_i128;

__device__ __forceinline__ void async_copy_b128(void* lds_dst, const void* gsrc) {
#if __has_builtin(__builtin_amdgcn_global_load_async_to_lds_b128)
  __builtin_amdgcn_global_load_async_to_lds_b128((gp_i128)gsrc, (lp_i128)lds_dst,
                                                 0, 0);
#else
  unsigned int loff = (unsigned int)(unsigned long long)lds_dst;  // LDS byte addr
  asm volatile("global_load_async_to_lds_b128 %0, %1, off"
               :: "v"(loff), "v"(gsrc) : "memory");
#endif
}

__device__ __forceinline__ void wait_async0() {
#if __has_builtin(__builtin_amdgcn_s_wait_asynccnt)
  __builtin_amdgcn_s_wait_asynccnt(0);
#else
  asm volatile("s_wait_asynccnt 0x0" ::: "memory");
#endif
}

// A-fragment 16x32 bf16 from LDS laid out [row][k], row-stride `stride` shorts.
// lanes 0-15 = rows M0-15 (K 0-15), lanes 16-31 same rows (K 16-31).
__device__ __forceinline__ v16bf load_frag_A(const unsigned short* p0, int stride, int lane) {
  Frag f;
  const int row = lane & 15;
  const int hi  = lane >> 4;
  const unsigned short* p = p0 + row * stride + 16 * hi;
  #pragma unroll
  for (int v = 0; v < 8; ++v) {
    const int kb = 2 * v;  // {0,2,...,14}
    f.u32[v] = *(const unsigned int*)(p + kb);
  }
  return f.bf;
}

// B-fragment 32x16 bf16 from LDS laid out [col][k].
// lanes 0-15 = cols N0-15 holding K0-15; lanes 16-31 same cols holding K16-31.
__device__ __forceinline__ v16bf load_frag_B(const unsigned short* p0, int stride, int lane) {
  Frag f;
  const int col = lane & 15;
  const int hi  = lane >> 4;
  const unsigned short* p = p0 + col * stride + 16 * hi;
  #pragma unroll
  for (int v = 0; v < 8; ++v)
    f.u32[v] = *(const unsigned int*)(p + 2 * v);
  return f.bf;
}

// Cooperative load of a [64 x ncols] fp32 weight matrix, transposed into
// LDS as bf16 [out_col][k] (stride 68 shorts).
__device__ __forceinline__ void load_WT(const float* __restrict__ W, int ncols,
                                        unsigned short (*lds)[68], int tid) {
  const int k  = tid >> 2;          // 0..63
  const int n0 = (tid & 3) * 16;    // 0,16,32,48
  const float* src = W + (size_t)k * ncols + n0;
  #pragma unroll
  for (int j = 0; j < 16; j += 4) {
    float4 f = *(const float4*)(src + j);
    lds[n0 + j + 0][k] = f2bf(f.x);
    lds[n0 + j + 1][k] = f2bf(f.y);
    lds[n0 + j + 2][k] = f2bf(f.z);
    lds[n0 + j + 3][k] = f2bf(f.w);
  }
}

// ---------------------------------------------------------------------------
// adj fp32 -> bf16 workspace copy (done once; reused by all 4 layers from L2)
// ---------------------------------------------------------------------------
__global__ __launch_bounds__(256) void convert_adj_kernel(
    const float* __restrict__ adj, unsigned short* __restrict__ adjb) {
  const size_t i = ((size_t)blockIdx.x * blockDim.x + threadIdx.x) * 8;
  float4 f0 = *(const float4*)(adj + i);
  float4 f1 = *(const float4*)(adj + i + 4);
  uint4 o;
  o.x = pack2(f0.x, f0.y);
  o.y = pack2(f0.z, f0.w);
  o.z = pack2(f1.x, f1.y);
  o.w = pack2(f1.z, f1.w);
  *(uint4*)(adjb + i) = o;
}

// ---------------------------------------------------------------------------
// out[row, n] = in[row, :64] @ W[:64, :64] + b[n] for a 64-row tile (no relu).
// ---------------------------------------------------------------------------
__global__ __launch_bounds__(256) void linear64_kernel(
    const float* __restrict__ in, const float* __restrict__ W,
    const float* __restrict__ bias, float* __restrict__ out) {
  __shared__ unsigned short lds_a[64][68];
  __shared__ unsigned short lds_w[64][68];

  const int tid = threadIdx.x, lane = tid & 31, wave = tid >> 5;
  const int rw = (wave >> 1) * 16;   // wave's 16-row group
  const int cw = (wave & 1) * 32;    // wave's 32-col group
  const size_t row0 = (size_t)blockIdx.x * 64;

  {  // stage A tile [64 x 64] -> bf16 [row][k]
    const int r = tid >> 2, d0 = (tid & 3) * 16;
    const float* src = in + (row0 + r) * kD + d0;
    unsigned int* dst = (unsigned int*)&lds_a[r][d0];
    #pragma unroll
    for (int j = 0; j < 16; j += 4) {
      float4 f = *(const float4*)(src + j);
      dst[j / 2 + 0] = pack2(f.x, f.y);
      dst[j / 2 + 1] = pack2(f.z, f.w);
    }
  }
  load_WT(W, kD, lds_w, tid);
  __syncthreads();

  v8f a0 = {}, a1 = {};
  #pragma unroll
  for (int kc = 0; kc < 64; kc += 32) {
    v16bf af  = load_frag_A(&lds_a[rw][kc], 68, lane);
    v16bf w0f = load_frag_B(&lds_w[cw][kc], 68, lane);
    v16bf w1f = load_frag_B(&lds_w[cw + 16][kc], 68, lane);
    a0 = wmma_bf16(af, w0f, a0);
    a1 = wmma_bf16(af, w1f, a1);
  }

  const int hi = lane >> 4, cl = lane & 15;
  const float bv0 = bias[cw + cl];
  const float bv1 = bias[cw + 16 + cl];
  #pragma unroll
  for (int r = 0; r < 8; ++r) {
    const int m = rw + r + 8 * hi;  // C/D layout: row = reg + 8*(lane>=16)
    out[(row0 + m) * kD + cw + cl]      = a0[r] + bv0;
    out[(row0 + m) * kD + cw + 16 + cl] = a1[r] + bv1;
  }
}

// ---------------------------------------------------------------------------
// One fused GIN layer for a (batch, 64-row) tile:
//   pooled = adj[b, rows, :] @ h[b] + (1+eps)*h[b, rows]
//   z      = relu(pooled @ W1 + b1)
//   h_out  = relu(z @ W2 + b2)
// adj chunks arrive via double-buffered async global->LDS copies (bf16).
// ---------------------------------------------------------------------------
__global__ __launch_bounds__(256) void gin_layer_kernel(
    const unsigned short* __restrict__ adjb,  // [B,N,N] bf16
    const float* __restrict__ h_in,
    const float* __restrict__ eps_l,
    const float* __restrict__ W1, const float* __restrict__ b1,
    const float* __restrict__ W2, const float* __restrict__ b2,
    float* __restrict__ h_out) {
  __shared__ alignas(16) unsigned short lds_adj[2][64][40];  // A chunk [row][k]
  __shared__ unsigned short lds_hT[2][64][36];               // B chunk [d][k]
  __shared__ unsigned short lds_p[64][68];                   // pooled (A, MLP1)
  __shared__ unsigned short lds_w[64][68];                   // W1^T / W2^T (B)
  __shared__ unsigned short lds_z[64][68];                   // z (A, MLP2)

  const int tid = threadIdx.x, lane = tid & 31, wave = tid >> 5;
  const int rw = (wave >> 1) * 16;
  const int cw = (wave & 1) * 32;
  const int bidx = blockIdx.y;
  const int row0 = blockIdx.x * 64;

  const unsigned short* adjb_b = adjb + ((size_t)bidx * kN + row0) * kN;
  const float* hin_b = h_in + (size_t)bidx * kN * kD;

  const int r_a = tid >> 2;          // adj stager: row 0..63
  const int k_a = (tid & 3) * 8;     //             8 consecutive k (16 B)
  const int k_h = tid >> 3;          // h stager:   k 0..31
  const int d_h = (tid & 7) * 8;     //             8 consecutive d

  // ---- phase 1: pooled = adj_tile @ h over K=512, 32-wide double-buffered --
  v8f acc0 = {}, acc1 = {};

  // prologue: stage chunk 0 into buffer 0
  async_copy_b128(&lds_adj[0][r_a][k_a], adjb_b + (size_t)r_a * kN + k_a);
  {
    const float* src = hin_b + (size_t)k_h * kD + d_h;
    float4 f0 = *(const float4*)(src);
    float4 f1 = *(const float4*)(src + 4);
    lds_hT[0][d_h + 0][k_h] = f2bf(f0.x);
    lds_hT[0][d_h + 1][k_h] = f2bf(f0.y);
    lds_hT[0][d_h + 2][k_h] = f2bf(f0.z);
    lds_hT[0][d_h + 3][k_h] = f2bf(f0.w);
    lds_hT[0][d_h + 4][k_h] = f2bf(f1.x);
    lds_hT[0][d_h + 5][k_h] = f2bf(f1.y);
    lds_hT[0][d_h + 6][k_h] = f2bf(f1.z);
    lds_hT[0][d_h + 7][k_h] = f2bf(f1.w);
  }

  for (int ck = 0; ck < kN / 32; ++ck) {
    const int p = ck & 1;
    wait_async0();        // this wave's async copy into buf[p] is complete
    __syncthreads();      // all waves' copies + stores done; prev compute done

    if (ck + 1 < kN / 32) {  // stage chunk ck+1 into the other buffer
      const int kc = (ck + 1) * 32;
      async_copy_b128(&lds_adj[1 - p][r_a][k_a],
                      adjb_b + (size_t)r_a * kN + kc + k_a);
      const float* src = hin_b + (size_t)(kc + k_h) * kD + d_h;
      float4 f0 = *(const float4*)(src);
      float4 f1 = *(const float4*)(src + 4);
      lds_hT[1 - p][d_h + 0][k_h] = f2bf(f0.x);
      lds_hT[1 - p][d_h + 1][k_h] = f2bf(f0.y);
      lds_hT[1 - p][d_h + 2][k_h] = f2bf(f0.z);
      lds_hT[1 - p][d_h + 3][k_h] = f2bf(f0.w);
      lds_hT[1 - p][d_h + 4][k_h] = f2bf(f1.x);
      lds_hT[1 - p][d_h + 5][k_h] = f2bf(f1.y);
      lds_hT[1 - p][d_h + 6][k_h] = f2bf(f1.z);
      lds_hT[1 - p][d_h + 7][k_h] = f2bf(f1.w);
    }

    v16bf af  = load_frag_A(&lds_adj[p][rw][0], 40, lane);
    v16bf bf0 = load_frag_B(&lds_hT[p][cw][0], 36, lane);
    v16bf bf1 = load_frag_B(&lds_hT[p][cw + 16][0], 36, lane);
    acc0 = wmma_bf16(af, bf0, acc0);
    acc1 = wmma_bf16(af, bf1, acc1);
  }

  // ---- self term + stash pooled (bf16) for MLP ----
  const float epse = 1.0f + *eps_l;
  const int hi = lane >> 4, cl = lane & 15;
  #pragma unroll
  for (int r = 0; r < 8; ++r) {
    const int m = rw + r + 8 * hi;
    const float* hr = hin_b + (size_t)(row0 + m) * kD;
    lds_p[m][cw + cl]      = f2bf(acc0[r] + epse * hr[cw + cl]);
    lds_p[m][cw + 16 + cl] = f2bf(acc1[r] + epse * hr[cw + 16 + cl]);
  }
  load_WT(W1, kH, lds_w, tid);
  __syncthreads();

  // ---- MLP layer 1: z = relu(pooled @ W1 + b1) ----
  v8f z0 = {}, z1 = {};
  #pragma unroll
  for (int kc = 0; kc < kD; kc += 32) {
    v16bf af  = load_frag_A(&lds_p[rw][kc], 68, lane);
    v16bf w0f = load_frag_B(&lds_w[cw][kc], 68, lane);
    v16bf w1f = load_frag_B(&lds_w[cw + 16][kc], 68, lane);
    z0 = wmma_bf16(af, w0f, z0);
    z1 = wmma_bf16(af, w1f, z1);
  }
  {
    const float bz0 = b1[cw + cl];
    const float bz1 = b1[cw + 16 + cl];
    #pragma unroll
    for (int r = 0; r < 8; ++r) {
      const int m = rw + r + 8 * hi;
      lds_z[m][cw + cl]      = f2bf(fmaxf(z0[r] + bz0, 0.0f));
      lds_z[m][cw + 16 + cl] = f2bf(fmaxf(z1[r] + bz1, 0.0f));
    }
  }
  __syncthreads();            // all waves done reading W1 / writing z
  load_WT(W2, kD, lds_w, tid);
  __syncthreads();

  // ---- MLP layer 2: h_out = relu(z @ W2 + b2) ----
  v8f o0 = {}, o1 = {};
  #pragma unroll
  for (int kc = 0; kc < kH; kc += 32) {
    v16bf af  = load_frag_A(&lds_z[rw][kc], 68, lane);
    v16bf w0f = load_frag_B(&lds_w[cw][kc], 68, lane);
    v16bf w1f = load_frag_B(&lds_w[cw + 16][kc], 68, lane);
    o0 = wmma_bf16(af, w0f, o0);
    o1 = wmma_bf16(af, w1f, o1);
  }
  {
    const float bo0 = b2[cw + cl];
    const float bo1 = b2[cw + 16 + cl];
    float* out_b = h_out + ((size_t)bidx * kN + row0) * kD;
    #pragma unroll
    for (int r = 0; r < 8; ++r) {
      const int m = rw + r + 8 * hi;
      out_b[(size_t)m * kD + cw + cl]      = fmaxf(o0[r] + bo0, 0.0f);
      out_b[(size_t)m * kD + cw + 16 + cl] = fmaxf(o1[r] + bo1, 0.0f);
    }
  }
}

// ---------------------------------------------------------------------------
// Deterministic global sum pool: g[b, off+d] = sum_n h[b,n,d]
// ---------------------------------------------------------------------------
__global__ void sum_pool_kernel(const float* __restrict__ h,
                                float* __restrict__ g, int off) {
  const int t = blockIdx.x * blockDim.x + threadIdx.x;
  if (t >= kB * kD) return;
  const int b = t / kD, d = t % kD;
  const float* p = h + (size_t)b * kN * kD + d;
  float s = 0.0f;
  for (int n = 0; n < kN; ++n) s += p[(size_t)n * kD];
  g[(size_t)b * kGD + off + d] = s;
}

// ---------------------------------------------------------------------------
// Classifier head: softmax(relu(g @ W1 + b1) @ W2 + b2). One block per batch.
// ---------------------------------------------------------------------------
__global__ __launch_bounds__(320) void head_kernel(
    const float* __restrict__ g, const float* __restrict__ W1,
    const float* __restrict__ b1, const float* __restrict__ W2,
    const float* __restrict__ b2, float* __restrict__ out) {
  __shared__ float sg[kGD];
  __shared__ float shg[kGD];
  __shared__ float slog[kC];
  const int b = blockIdx.x, t = threadIdx.x;
  sg[t] = g[(size_t)b * kGD + t];
  __syncthreads();
  float acc = b1[t];
  for (int k = 0; k < kGD; ++k) acc = fmaf(sg[k], W1[(size_t)k * kGD + t], acc);
  shg[t] = fmaxf(acc, 0.0f);
  __syncthreads();
  if (t < kC) {
    float a = b2[t];
    for (int k = 0; k < kGD; ++k) a = fmaf(shg[k], W2[(size_t)k * kC + t], a);
    slog[t] = a;
  }
  __syncthreads();
  if (t == 0) {
    float mx = slog[0];
    for (int i = 1; i < kC; ++i) mx = fmaxf(mx, slog[i]);
    float e[kC], s = 0.0f;
    for (int i = 0; i < kC; ++i) { e[i] = __expf(slog[i] - mx); s += e[i]; }
    const float inv = 1.0f / s;
    for (int i = 0; i < kC; ++i) out[(size_t)b * kC + i] = e[i] * inv;
  }
}

extern "C" void kernel_launch(void* const* d_in, const int* in_sizes, int n_in,
                              void* d_out, int out_size, void* d_ws, size_t ws_size,
                              hipStream_t stream) {
  (void)in_sizes; (void)n_in; (void)out_size; (void)ws_size;
  const float* adj    = (const float*)d_in[0];
  const float* x      = (const float*)d_in[1];
  const float* firstW = (const float*)d_in[2];
  const float* firstB = (const float*)d_in[3];
  const float* eps    = (const float*)d_in[4];
  const float* W1     = (const float*)d_in[5];
  const float* b1     = (const float*)d_in[6];
  const float* W2     = (const float*)d_in[7];
  const float* b2     = (const float*)d_in[8];
  const float* l1W    = (const float*)d_in[9];
  const float* l1b    = (const float*)d_in[10];
  const float* l2W    = (const float*)d_in[11];
  const float* l2b    = (const float*)d_in[12];
  float* out = (float*)d_out;

  // workspace: adj bf16 copy + two [B,N,D] fp32 ping-pong buffers + g [B,320]
  unsigned short* adjb = (unsigned short*)d_ws;            // B*N*N bf16 (33.5 MB)
  float* hA = (float*)(adjb + (size_t)kB * kN * kN);
  float* hB = hA + (size_t)kB * kN * kD;
  float* g  = hB + (size_t)kB * kN * kD;

  const size_t adj_elems = (size_t)kB * kN * kN;
  convert_adj_kernel<<<(unsigned)(adj_elems / (8 * 256)), 256, 0, stream>>>(adj, adjb);

  const int row_tiles = (kB * kN) / 64;  // 512 blocks
  linear64_kernel<<<row_tiles, 256, 0, stream>>>(x, firstW, firstB, hA);
  sum_pool_kernel<<<(kB * kD + 255) / 256, 256, 0, stream>>>(hA, g, 0);

  float* cur = hA;
  float* nxt = hB;
  for (int l = 0; l < kL; ++l) {
    gin_layer_kernel<<<dim3(kN / 64, kB), 256, 0, stream>>>(
        adjb, cur, eps + l, W1 + (size_t)l * kD * kH, b1 + (size_t)l * kH,
        W2 + (size_t)l * kH * kD, b2 + (size_t)l * kD, nxt);
    sum_pool_kernel<<<(kB * kD + 255) / 256, 256, 0, stream>>>(nxt, g, (l + 1) * kD);
    float* t = cur; cur = nxt; nxt = t;
  }
  head_kernel<<<kB, 320, 0, stream>>>(g, l1W, l1b, l2W, l2b, out);
}
